// NisuyNN_90434831384984
// MI455X (gfx1250) — matrix operand order repaired and checked by reading, not compiled, not run on hardware
//
#include <hip/hip_runtime.h>
#include <hip/hip_bf16.h>
#include <stdint.h>

// Problem constants (from reference)
#define DIMD     128
#define NNODE    32
#define BATCH    4
#define MROWS    4096      // N*N*B rows through the MLP
#define PI_ITERS 50
#define SLOPE    0.01f

typedef __bf16 bf16t;
typedef __attribute__((ext_vector_type(16))) __bf16 v16bf;
typedef __attribute__((ext_vector_type(8)))  float  v8f;

union Frag { v16bf v; uint4 q[2]; };

// ---------------------------------------------------------------------------
// 1) Build MLP input X (4096 x 256) bf16, replicating torch stack/view order.
//    row r = ((s*32 + t)*4 + b'); col c; f = b'*256+c -> i2=f>>9, b=(f>>7)&3, j=f&127
// ---------------------------------------------------------------------------
__global__ void build_x_kernel(const float* __restrict__ emb, bf16t* __restrict__ X) {
    int r  = blockIdx.x;          // 0..4095
    int c  = threadIdx.x;         // 0..255
    int bq = r & 3;
    int st = r >> 2;
    int t  = st & 31;
    int s  = st >> 5;
    int f  = bq * 256 + c;
    int i2 = f >> 9;
    int b  = (f >> 7) & 3;
    int j  = f & 127;
    int node = i2 ? t : s;
    float v = emb[(b * NNODE + node) * DIMD + j];
    X[(size_t)r * 256 + c] = (bf16t)v;
}

// ---------------------------------------------------------------------------
// 2a) Transpose-convert weights: W (K x N, f32, row-major) -> Wt (N x K, bf16)
//     so GEMM B-tiles are K-contiguous (TDM-friendly 2D tiles).
// ---------------------------------------------------------------------------
__global__ __launch_bounds__(256) void convT_kernel(const float* __restrict__ W,
                                                    bf16t* __restrict__ Wt,
                                                    int K, int N) {
    __shared__ float tile[32][33];
    int nB = blockIdx.x * 32;
    int kB = blockIdx.y * 32;
    int tx = threadIdx.x;   // 0..31
    int ty = threadIdx.y;   // 0..7
#pragma unroll
    for (int i = 0; i < 4; ++i)
        tile[ty + i * 8][tx] = W[(size_t)(kB + ty + i * 8) * N + nB + tx];
    __syncthreads();
#pragma unroll
    for (int i = 0; i < 4; ++i)
        Wt[(size_t)(nB + ty + i * 8) * K + kB + tx] = (bf16t)tile[tx][ty + i * 8];
}

// ---------------------------------------------------------------------------
// TDM helper: 2D tile load (tile = 128 rows x 64 bf16, row stride = K elems)
// Descriptor packing per CDNA5 ISA.
// ---------------------------------------------------------------------------
#if __has_builtin(__builtin_amdgcn_tensor_load_to_lds)
#define HAS_TDM 1
typedef __attribute__((ext_vector_type(4))) unsigned int u32x4;
typedef __attribute__((ext_vector_type(8))) int i32x8;
typedef __attribute__((ext_vector_type(4))) int i32x4;

__device__ __forceinline__ void tdm_load_tile_128x64(unsigned ldsOff,
                                                     const bf16t* gptr,
                                                     unsigned tensorDim0,   // row length (elems)
                                                     unsigned tensorDim1,   // #rows in tensor
                                                     unsigned long long rowStride) // elems
{
    unsigned long long ga = (unsigned long long)(const void*)gptr;
    u32x4 g0;
    g0[0] = 1u;                                        // count=1, user descriptor
    g0[1] = ldsOff;                                    // lds_addr
    g0[2] = (unsigned)(ga & 0xFFFFFFFFu);              // global_addr[31:0]
    g0[3] = (unsigned)((ga >> 32) & 0x01FFFFFFu)       // global_addr[56:32]
          | (2u << 30);                                // type = 2 ("image")
    i32x8 g1;
    g1[0] = (int)(1u << 16);                           // data_size=1 (2 bytes); wg_mask=0
    g1[1] = (int)((tensorDim0 & 0xFFFFu) << 16);       // tensor_dim0[15:0]
    g1[2] = (int)(((tensorDim0 >> 16) & 0xFFFFu)       // tensor_dim0[31:16]
          | ((tensorDim1 & 0xFFFFu) << 16));           // tensor_dim1[15:0]
    g1[3] = (int)(((tensorDim1 >> 16) & 0xFFFFu)       // tensor_dim1[31:16]
          | (64u << 16));                              // tile_dim0 = 64 elems
    g1[4] = (int)128u;                                 // tile_dim1 = 128 rows; tile_dim2=0
    g1[5] = (int)(unsigned)(rowStride & 0xFFFFFFFFull);        // dim0_stride[31:0]
    g1[6] = (int)(unsigned)((rowStride >> 32) & 0xFFFFull);    // dim0_stride[47:32]
    g1[7] = 0;                                         // dim1_stride unused (2D)
    i32x4 z4 = {0, 0, 0, 0};
#if defined(__clang_major__) && (__clang_major__ >= 23)
    i32x8 z8 = {0, 0, 0, 0, 0, 0, 0, 0};
    __builtin_amdgcn_tensor_load_to_lds(g0, g1, z4, z4, z8, 0);
#else
    __builtin_amdgcn_tensor_load_to_lds(g0, g1, z4, z4, 0);
#endif
}
#else
#define HAS_TDM 0
#endif

// ---------------------------------------------------------------------------
// 2b) GEMM: C = act(A @ Wt^T + bias)
//     A : M x K bf16 row-major, Wt : N x K bf16 row-major (B pre-transposed)
//     Block tile 128x128, K-step 64, double-buffered TDM pipeline:
//       issue DMA for tile k+1, compute tile k from LDS, drain tensorcnt,
//       barrier, swap. 8 waves (2 N x 4 M), wave tile 32x64, 2x4 fragments
//       of v_wmma_f32_16x16x32_bf16.
//     mode 0: LeakyReLU -> bf16 Cb ; mode 1: sigmoid*mult+add -> f32 Cf
// ---------------------------------------------------------------------------
__global__ __launch_bounds__(256) void gemm_wmma_bf16(
    const bf16t* __restrict__ A, const bf16t* __restrict__ Bt,
    const float* __restrict__ bias, bf16t* __restrict__ Cb, float* __restrict__ Cf,
    int M, int N, int K, int mode,
    const float* __restrict__ multp, const float* __restrict__ addp)
{
    __shared__ __align__(16) bf16t As[2][128 * 64];   // 2 x 16 KB
    __shared__ __align__(16) bf16t Bs[2][128 * 64];   // 2 x 16 KB

    const int tid   = threadIdx.x;
    const int wave  = tid >> 5;
    const int lane  = tid & 31;
    const int half  = lane >> 4;     // 0 or 1
    const int lr    = lane & 15;
    const int waveM = wave & 3;      // 4 waves along M
    const int waveN = wave >> 2;     // 2 waves along N
    const int mBlk  = blockIdx.y * 128;
    const int nBlk  = blockIdx.x * 128;
    const int mLoc  = waveM * 32;
    const int nLoc  = waveN * 64;

    const v8f vzero = {0.f, 0.f, 0.f, 0.f, 0.f, 0.f, 0.f, 0.f};
    v8f acc[2][4];
#pragma unroll
    for (int mi = 0; mi < 2; ++mi)
#pragma unroll
        for (int ni = 0; ni < 4; ++ni) acc[mi][ni] = vzero;

#if HAS_TDM
    // Prologue: DMA tile 0 into buffer 0.
    if (wave == 0) {
        tdm_load_tile_128x64((unsigned)(size_t)(&As[0][0]),
                             A + (size_t)mBlk * K,
                             (unsigned)K, (unsigned)M, (unsigned long long)K);
        tdm_load_tile_128x64((unsigned)(size_t)(&Bs[0][0]),
                             Bt + (size_t)nBlk * K,
                             (unsigned)K, (unsigned)N, (unsigned long long)K);
        __builtin_amdgcn_s_wait_tensorcnt(0);
    }
    __syncthreads();
#endif

    int cur = 0;
    for (int kb = 0; kb < K; kb += 64) {
#if HAS_TDM
        // Kick off the DMA for the next K-tile into the other buffer; the
        // barrier that ended the previous iteration guarantees nobody still
        // reads it. The transfer overlaps with the WMMA work below.
        if (wave == 0 && kb + 64 < K) {
            tdm_load_tile_128x64((unsigned)(size_t)(&As[cur ^ 1][0]),
                                 A + (size_t)mBlk * K + kb + 64,
                                 (unsigned)K, (unsigned)M, (unsigned long long)K);
            tdm_load_tile_128x64((unsigned)(size_t)(&Bs[cur ^ 1][0]),
                                 Bt + (size_t)nBlk * K + kb + 64,
                                 (unsigned)K, (unsigned)N, (unsigned long long)K);
        }
#else
        __syncthreads();
        for (int i = tid; i < 128 * 8; i += 256) {   // 8 uint4 per 64-elem row
            int row = i >> 3, qc = i & 7;
            *(uint4*)&As[cur][row * 64 + qc * 8] =
                *(const uint4*)&A[(size_t)(mBlk + row) * K + kb + qc * 8];
            *(uint4*)&Bs[cur][row * 64 + qc * 8] =
                *(const uint4*)&Bt[(size_t)(nBlk + row) * K + kb + qc * 8];
        }
        __syncthreads();
#endif

        const bf16t* __restrict__ Asb = &As[cur][0];
        const bf16t* __restrict__ Bsb = &Bs[cur][0];
#pragma unroll
        for (int kk = 0; kk < 2; ++kk) {
            Frag af[2], bfr[4];
#pragma unroll
            for (int mi = 0; mi < 2; ++mi) {
                int row = mLoc + mi * 16 + lr;
                int k0  = kk * 32 + half * 8;           // A 16-bit layout: half-wave K split
                af[mi].q[0] = *(const uint4*)&Asb[row * 64 + k0];
                af[mi].q[1] = *(const uint4*)&Asb[row * 64 + k0 + 16];
            }
#pragma unroll
            for (int ni = 0; ni < 4; ++ni) {
                int col = nLoc + ni * 16 + lr;
                int k0  = kk * 32 + half * 8;           // B: lane=col, same half-wave K split as A
                bfr[ni].q[0] = *(const uint4*)&Bsb[col * 64 + k0];
                bfr[ni].q[1] = *(const uint4*)&Bsb[col * 64 + k0 + 16];
            }
#pragma unroll
            for (int mi = 0; mi < 2; ++mi)
#pragma unroll
                for (int ni = 0; ni < 4; ++ni)
                    acc[mi][ni] = __builtin_amdgcn_wmma_f32_16x16x32_bf16(
                        false, af[mi].v, false, bfr[ni].v,
                        (short)0, acc[mi][ni], false, false);
        }

#if HAS_TDM
        // Drain the in-flight DMA, then publish the fresh buffer to all waves.
        if (wave == 0) __builtin_amdgcn_s_wait_tensorcnt(0);
        __syncthreads();
        cur ^= 1;
#endif
    }

    // Epilogue. C fragment layout: lane = (half,lr) -> col, VGPR v -> row v+8*half.
    float mc = 1.f, ac = 0.f;
    if (mode) { mc = multp[0]; ac = addp[0]; }
#pragma unroll
    for (int ni = 0; ni < 4; ++ni) {
        int col = nBlk + nLoc + ni * 16 + lr;
        float bv = bias[col];
#pragma unroll
        for (int mi = 0; mi < 2; ++mi) {
#pragma unroll
            for (int v = 0; v < 8; ++v) {
                int row = mBlk + mLoc + mi * 16 + half * 8 + v;
                float x = acc[mi][ni][v] + bv;
                if (mode == 0) {
                    x = (x > 0.f) ? x : SLOPE * x;
                    Cb[(size_t)row * N + col] = (bf16t)x;
                } else {
                    x = 1.f / (1.f + __expf(-x));
                    Cf[(size_t)row * N + col] = x * mc + ac;
                }
            }
        }
    }
}

// ---------------------------------------------------------------------------
// 3) Power iteration: 4096 matrices of 32x32, 50 steps, one wave per matrix.
// ---------------------------------------------------------------------------
__global__ __launch_bounds__(32) void power_iter_kernel(const float* __restrict__ P,
                                                        float* __restrict__ eig) {
    int k = blockIdx.x;
    int i = threadIdx.x;          // lane = row
    float m[32];
#pragma unroll
    for (int j = 0; j < 32; ++j) m[j] = P[(size_t)k * 1024 + i * 32 + j];
    float b = 1.0f / 32.0f;
    for (int it = 0; it < PI_ITERS; ++it) {
        float v = 0.f;
#pragma unroll
        for (int j = 0; j < 32; ++j) v = fmaf(m[j], __shfl(b, j, 32), v);
        float s = v * v;
#pragma unroll
        for (int off = 16; off > 0; off >>= 1) s += __shfl_xor(s, off, 32);
        b = v * rsqrtf(s);
    }
    eig[(size_t)k * 32 + i] = b;
}

// ---------------------------------------------------------------------------
// 4) deltas[b,i] = sum_{s,t} (eig[s,t,b,i]/eig[s,t,b,s]) * T[b,s,t]
// ---------------------------------------------------------------------------
__global__ __launch_bounds__(32) void deltas_kernel(const float* __restrict__ eig,
                                                    const float* __restrict__ Ts,
                                                    float* __restrict__ out) {
    int b = blockIdx.x;
    int i = threadIdx.x;
    float acc = 0.f;
    for (int st = 0; st < NNODE * NNODE; ++st) {
        int s = st >> 5;
        float v   = eig[((size_t)st * BATCH + b) * 32 + i];
        float src = __shfl(v, s, 32);
        float tv  = Ts[(size_t)b * 1024 + st];
        acc += (v / src) * tv;
    }
    out[b * NNODE + i] = acc;
}

// ---------------------------------------------------------------------------
// Host-side orchestration
// ---------------------------------------------------------------------------
extern "C" void kernel_launch(void* const* d_in, const int* in_sizes, int n_in,
                              void* d_out, int out_size, void* d_ws, size_t ws_size,
                              hipStream_t stream) {
    const float* emb  = (const float*)d_in[0];
    // d_in[1] (batch_graphs_embeddings) unused, as in reference
    const float* Ts   = (const float*)d_in[2];
    const float* mult = (const float*)d_in[3];
    const float* addc = (const float*)d_in[4];
    const float* W[6]   = {(const float*)d_in[5],  (const float*)d_in[7],
                           (const float*)d_in[9],  (const float*)d_in[11],
                           (const float*)d_in[13], (const float*)d_in[15]};
    const float* bia[6] = {(const float*)d_in[6],  (const float*)d_in[8],
                           (const float*)d_in[10], (const float*)d_in[12],
                           (const float*)d_in[14], (const float*)d_in[16]};

    // Workspace carve-up (~118 MB)
    char*  ws   = (char*)d_ws;
    bf16t* actA = (bf16t*)(ws);                          // 32 MB: activations ping
    bf16t* actB = (bf16t*)(ws + (size_t)33554432);       // 32 MB: activations pong
    bf16t* wtb  = (bf16t*)(ws + (size_t)67108864);       // 32 MB: bf16 transposed weights
    float* pol  = (float*)(ws + (size_t)100663296);      // 16 MB: policies (4096x1024 f32)
    float* eig  = (float*)(ws + (size_t)117440512);      // 0.5 MB: eigenvectors
    float* out  = (float*)d_out;

    build_x_kernel<<<MROWS, 256, 0, stream>>>(emb, actA);

    const int Ks[6] = {2 * DIMD, 4096, 4096, 4096, 4096, 4096};
    const int Ns[6] = {4096, 4096, 4096, 4096, 4096, NNODE * NNODE};
    bf16t* cur = actA;
    bf16t* nxt = actB;
    for (int l = 0; l < 6; ++l) {
        int Kd = Ks[l], Nd = Ns[l];
        convT_kernel<<<dim3(Nd / 32, Kd / 32), dim3(32, 8), 0, stream>>>(W[l], wtb, Kd, Nd);
        int mode = (l == 5) ? 1 : 0;
        gemm_wmma_bf16<<<dim3(Nd / 128, MROWS / 128), 256, 0, stream>>>(
            cur, wtb, bia[l], nxt, pol, MROWS, Nd, Kd, mode, mult, addc);
        bf16t* t = cur; cur = nxt; nxt = t;
    }

    power_iter_kernel<<<MROWS, 32, 0, stream>>>(pol, eig);
    deltas_kernel<<<BATCH, NNODE, 0, stream>>>(eig, Ts, out);
}
